// FBSNN_74921409511445
// MI455X (gfx1250) — compile-verified, matrix operand order checked by reading, not run on hardware
//
#include <hip/hip_runtime.h>

typedef __attribute__((ext_vector_type(16))) _Float16 v16h;
typedef __attribute__((ext_vector_type(8)))  _Float16 v8h;
typedef __attribute__((ext_vector_type(8)))  float    v8f;

#define MPATH   2048
#define NSTEP   20
#define NT      (NSTEP + 1)            // 21
#define NPTS    (MPATH * NT)           // 43008
#define HIDDIM  256
#define R_RATE  0.05f
#define SIGMA_V 0.4f
#define KSTR    1.0f
#define TMAT_V  1.0f

#define ROWS 32          // points per workgroup
#define HS   264         // LDS row stride in halves (256 + 8 pad, keeps 16B align)

struct BiasPtrs { const float* b[9]; };

// ---------------------------------------------------------------------------
// Pack one 256x256 fp32 weight matrix (row = input k, col = output n) into
// f16 WMMA B-fragments.  Fragment index: (ct, kb, lane) -> v16h where
//   n  = lane & 15  (output column within 16-wide tile ct)
//   halves 0..7  : k = kb*32 + (lane>>4)*8 + j
//   halves 8..15 : k = kb*32 + (lane>>4)*8 + 16 + j
// matching the ISA 16-bit 32x16 B-matrix VGPR layout.
// ---------------------------------------------------------------------------
__global__ void pack_w(const float* __restrict__ W, v16h* __restrict__ dst) {
    int tid  = blockIdx.x * blockDim.x + threadIdx.x;   // 0..4095
    int ct   = tid >> 8;                                // 16 col tiles
    int kb   = (tid >> 5) & 7;                          // 8 k-blocks
    int lane = tid & 31;
    int n    = lane & 15;
    int kh   = (lane >> 4) * 8;
    int col  = ct * 16 + n;
    v16h v;
#pragma unroll
    for (int j = 0; j < 8; ++j) {
        int k0 = kb * 32 + kh + j;
        int k1 = kb * 32 + kh + 16 + j;
        v[j]     = (_Float16)W[k0 * HIDDIM + col];
        v[j + 8] = (_Float16)W[k1 * HIDDIM + col];
    }
    dst[(ct * 8 + kb) * 32 + lane] = v;
}

// ---------------------------------------------------------------------------
// GBM Euler X-path (independent of the network).  Also writes the X block of
// d_out and zeroes the loss slot.
// ---------------------------------------------------------------------------
__global__ void xpath_kernel(const float* __restrict__ tarr,
                             const float* __restrict__ Wp,
                             const float* __restrict__ Xi,
                             float* __restrict__ Xbuf,
                             float* __restrict__ out) {
    int m = blockIdx.x * blockDim.x + threadIdx.x;
    if (m == 0) out[0] = 0.0f;
    if (m >= MPATH) return;
    float X  = Xi[m];
    float tp = tarr[m * NT];
    float Wc = Wp[m * NT];
    Xbuf[m * NT] = X;
    out[1 + m * NT] = X;
    for (int n = 1; n <= NSTEP; ++n) {
        float tn = tarr[m * NT + n];
        float Wn = Wp[m * NT + n];
        X = X + R_RATE * X * (tn - tp) + SIGMA_V * X * (Wn - Wc);
        tp = tn; Wc = Wn;
        Xbuf[m * NT + n] = X;
        out[1 + m * NT + n] = X;
    }
}

// ---------------------------------------------------------------------------
// Persistent-tile MLP: each workgroup evaluates u = MLP(t, x) for 32 points.
// 8 waves; each wave owns BOTH 16-row tiles x 2 col-tiles (32 cols), so every
// B fragment loaded from L2 is reused across two WMMAs (A reuse comes from
// LDS, which is nearly free).  9 hidden layers via v_wmma_f32_16x16x32_f16.
// Activations use the hardware transcendental (__sinf -> v_sin_f32, TRANS
// unit, co-executes with WMMA) instead of the ~30-op precise sinf chain.
// ---------------------------------------------------------------------------
__global__ __launch_bounds__(256) void mlp_eval(
    const float* __restrict__ tarr,
    const float* __restrict__ Xbuf,
    const float* __restrict__ W0,   // [2,256] fp32
    const float* __restrict__ b0,   // [256]
    const v16h*  __restrict__ Bpack,// packed f16 weights, 4096 v16h per layer
    BiasPtrs bp,                    // b1..b9
    const float* __restrict__ Wo,   // [256]
    const float* __restrict__ bo,   // [1]
    float* __restrict__ Ybuf,
    float* __restrict__ out) {

    __shared__ _Float16 __attribute__((aligned(16))) Hbuf[2][ROWS * HS];
    __shared__ float red[ROWS][8];

    const int tid = threadIdx.x;
    const int q0  = blockIdx.x * ROWS;

    // ---- layer 0: [t,x] (2) -> 256, sin activation (VALU, too skinny for WMMA)
    {
        int   p  = tid >> 3;            // point 0..31
        int   q  = q0 + p;
        float tv = tarr[q];
        float xv = Xbuf[q];
        int   cb = (tid & 7) * 32;
#pragma unroll 8
        for (int j = 0; j < 32; ++j) {
            int   c = cb + j;
            float z = tv * W0[c] + xv * W0[HIDDIM + c] + b0[c];
            Hbuf[0][p * HS + c] = (_Float16)__sinf(z);
        }
    }
    __syncthreads();

    const int wave = tid >> 5;
    const int lane = tid & 31;
    const int ct0  = wave * 2;          // two col tiles per wave
    const int ct1  = wave * 2 + 1;
    const int lr   = lane & 15;
    const int lh   = lane >> 4;

    int cur = 0;
    for (int l = 0; l < 9; ++l) {
        const _Float16* Hin  = Hbuf[cur];
        _Float16*       Hout = Hbuf[cur ^ 1];
        const v16h*     Bl   = Bpack + l * 4096;

        v8f a00 = {}, a01 = {}, a10 = {}, a11 = {};
        const int arow0 = lr * HS;              // row tile 0: rows 0..15
        const int arow1 = (16 + lr) * HS;       // row tile 1: rows 16..31

#pragma unroll
        for (int kb = 0; kb < 8; ++kb) {
            // A fragments: ISA 16-bit 16x32 layout -> two contiguous 16B LDS
            // loads per row tile
            int k0 = kb * 32 + lh * 8;
            v8h p0 = *(const v8h*)(Hin + arow0 + k0);
            v8h p1 = *(const v8h*)(Hin + arow0 + k0 + 16);
            v8h p2 = *(const v8h*)(Hin + arow1 + k0);
            v8h p3 = *(const v8h*)(Hin + arow1 + k0 + 16);
            v16h fa0 = __builtin_shufflevector(p0, p1, 0,1,2,3,4,5,6,7,
                                                       8,9,10,11,12,13,14,15);
            v16h fa1 = __builtin_shufflevector(p2, p3, 0,1,2,3,4,5,6,7,
                                                       8,9,10,11,12,13,14,15);
            // B fragments: one 32B contiguous global load each, reused twice
            v16h fb0 = Bl[(ct0 * 8 + kb) * 32 + lane];
            v16h fb1 = Bl[(ct1 * 8 + kb) * 32 + lane];
            a00 = __builtin_amdgcn_wmma_f32_16x16x32_f16(false, fa0, false, fb0,
                                                         (short)0, a00, false, false);
            a01 = __builtin_amdgcn_wmma_f32_16x16x32_f16(false, fa0, false, fb1,
                                                         (short)0, a01, false, false);
            a10 = __builtin_amdgcn_wmma_f32_16x16x32_f16(false, fa1, false, fb0,
                                                         (short)0, a10, false, false);
            a11 = __builtin_amdgcn_wmma_f32_16x16x32_f16(false, fa1, false, fb1,
                                                         (short)0, a11, false, false);
        }

        // epilogue: bias + sin, scatter into Hout per C/D layout
        // (lanes 0-15: N = lane, M = r; lanes 16-31: N = lane-16, M = 8 + r)
        const float* bias = bp.b[l];
        const int mb  = lh * 8;
        const int c0  = ct0 * 16 + lr;
        const int c1  = ct1 * 16 + lr;
        const float bv0 = bias[c0], bv1 = bias[c1];
#pragma unroll
        for (int r = 0; r < 8; ++r) {
            int row0 = (mb + r) * HS;
            int row1 = (16 + mb + r) * HS;
            Hout[row0 + c0] = (_Float16)__sinf(a00[r] + bv0);
            Hout[row0 + c1] = (_Float16)__sinf(a01[r] + bv1);
            Hout[row1 + c0] = (_Float16)__sinf(a10[r] + bv0);
            Hout[row1 + c1] = (_Float16)__sinf(a11[r] + bv1);
        }
        __syncthreads();
        cur ^= 1;
    }

    // ---- output layer: 256 -> 1 (dot product, 8 threads per point)
    {
        const _Float16* Hf = Hbuf[cur];
        int p  = tid >> 3;
        int cb = (tid & 7) * 32;
        float s = 0.0f;
#pragma unroll 8
        for (int j = 0; j < 32; ++j) {
            int c = cb + j;
            s += (float)Hf[p * HS + c] * Wo[c];
        }
        red[p][tid & 7] = s;
    }
    __syncthreads();
    if (tid < ROWS) {
        float y = bo[0];
#pragma unroll
        for (int s = 0; s < 8; ++s) y += red[tid][s];
        int q = q0 + tid;
        Ybuf[q] = y;
        out[1 + NPTS + q] = y;
        if (q == 0) out[1 + 2 * NPTS] = y;   // Y[0,0,0]
    }
}

// ---------------------------------------------------------------------------
// Loss: sum over n>=1 of (bs_call(X, T - t) - Y)^2, deterministic 2-stage.
// ---------------------------------------------------------------------------
__device__ __forceinline__ float ncdf(float d) {
    return 0.5f * erfcf(-d * 0.70710678118654752f);
}

__global__ void loss_partial(const float* __restrict__ tarr,
                             const float* __restrict__ Xbuf,
                             const float* __restrict__ Ybuf,
                             float* __restrict__ partial) {
    __shared__ float sm[256];
    int tid = threadIdx.x;
    int q   = blockIdx.x * 256 + tid;
    float v = 0.0f;
    if (q < NPTS) {
        int n = q % NT;
        if (n != 0) {
            float X    = Xbuf[q];
            float tau  = TMAT_V - tarr[q];
            float taus = fmaxf(tau, 1e-8f);
            float st   = sqrtf(taus);
            float d1   = (logf(X / KSTR) + (R_RATE + 0.5f * SIGMA_V * SIGMA_V) * taus)
                         / (SIGMA_V * st);
            float d2   = d1 - SIGMA_V * st;
            float c    = X * ncdf(d1) - KSTR * expf(-R_RATE * taus) * ncdf(d2);
            float pay  = fmaxf(X - KSTR, 0.0f);
            float ex   = (tau > 0.0f) ? c : pay;
            float d    = ex - Ybuf[q];
            v = d * d;
        }
    }
    sm[tid] = v;
    __syncthreads();
    for (int s = 128; s > 0; s >>= 1) {
        if (tid < s) sm[tid] += sm[tid + s];
        __syncthreads();
    }
    if (tid == 0) partial[blockIdx.x] = sm[0];
}

__global__ void loss_final(const float* __restrict__ partial, int nb,
                           float* __restrict__ out) {
    if (threadIdx.x == 0) {
        float s = 0.0f;
        for (int i = 0; i < nb; ++i) s += partial[i];
        out[0] = s / (float)(NSTEP + 1);   // N + LAMBDA = 21
    }
}

// ---------------------------------------------------------------------------
extern "C" void kernel_launch(void* const* d_in, const int* in_sizes, int n_in,
                              void* d_out, int out_size, void* d_ws, size_t ws_size,
                              hipStream_t stream) {
    (void)in_sizes; (void)n_in; (void)out_size; (void)ws_size;

    // d_in order: t, W, Xi, then params leaves: (W0,b0),(W1,b1)...(W9,b9),(Wo,bo)
    const float* tarr = (const float*)d_in[0];
    const float* Wp   = (const float*)d_in[1];
    const float* Xi   = (const float*)d_in[2];
    const float* W0   = (const float*)d_in[3];
    const float* b0   = (const float*)d_in[4];
    const float* Wo   = (const float*)d_in[23];
    const float* bo   = (const float*)d_in[24];

    char*  ws      = (char*)d_ws;
    v16h*  Bpack   = (v16h*)ws;                      // 9*4096*32B = 1,179,648 B
    float* Xbuf    = (float*)(ws + 1179648);         // 43008 f32
    float* Ybuf    = (float*)(ws + 1351680);         // 43008 f32
    float* partial = (float*)(ws + 1523712);         // 168 f32
    float* out     = (float*)d_out;

    // pack the 9 hidden 256x256 weight matrices to f16 WMMA B-fragments
    for (int l = 0; l < 9; ++l) {
        const float* Wl = (const float*)d_in[3 + 2 * (l + 1)];
        pack_w<<<16, 256, 0, stream>>>(Wl, Bpack + l * 4096);
    }

    xpath_kernel<<<(MPATH + 255) / 256, 256, 0, stream>>>(tarr, Wp, Xi, Xbuf, out);

    BiasPtrs bp;
    for (int l = 0; l < 9; ++l) bp.b[l] = (const float*)d_in[4 + 2 * (l + 1)];

    mlp_eval<<<NPTS / ROWS, 256, 0, stream>>>(tarr, Xbuf, W0, b0, Bpack, bp,
                                              Wo, bo, Ybuf, out);

    const int nb = (NPTS + 255) / 256;   // 168
    loss_partial<<<nb, 256, 0, stream>>>(tarr, Xbuf, Ybuf, partial);
    loss_final<<<1, 32, 0, stream>>>(partial, nb, out);
}
